// Attention_83648783057141
// MI455X (gfx1250) — matrix operand order repaired and checked by reading, compiled, not run on hardware
//
#include <hip/hip_runtime.h>
#include <hip/hip_bf16.h>
#include <math.h>

#define BATCH 16
#define SEQ   4096
#define HDIM  512
#define EDIM  1024   // 2*H

#define MTILE    64   // encoder rows per workgroup in the WMMA kernel
#define LG_BLOCK 128  // 4 wave32 waves, each owns 16 rows
#define NBLK     4    // n-tiles processed together per wave (register blocking)

typedef __attribute__((ext_vector_type(16))) __bf16 v16bf;
typedef __attribute__((ext_vector_type(8)))  float  v8f;

// round-to-nearest-even fp32 -> bf16 (bit trick, deterministic)
static __device__ __forceinline__ unsigned short f2bf(float x) {
    unsigned u = __float_as_uint(x);
    unsigned r = u + 0x7FFFu + ((u >> 16) & 1u);
    return (unsigned short)(r >> 16);
}

// ---------------------------------------------------------------------------
// hid_proj[b,h] = sum_k hidden[b,k] * Wh[k,h] + b_attn[h]   (Wh = W_attn rows 0..511)
__global__ void k_hidproj(const float* __restrict__ hidden,
                          const float* __restrict__ W,
                          const float* __restrict__ bias,
                          float* __restrict__ hidp) {
    int gid = blockIdx.x * blockDim.x + threadIdx.x;   // 0..8191
    int b = gid >> 9;
    int h = gid & (HDIM - 1);
    float s = bias[h];
    const float* hv = hidden + (size_t)b * HDIM;
#pragma unroll 4
    for (int k = 0; k < HDIM; ++k) s += hv[k] * W[(size_t)k * HDIM + h];
    hidp[gid] = s;
}

// ---------------------------------------------------------------------------
// WeT[h][e] = bf16( W_attn[H + e][h] )   -- K(=e)-contiguous per column h,
// which matches the WMMA B-fragment layout (lane half h -> K 8h..8h+7 and 16+8h..23+8h).
__global__ void k_wet(const float* __restrict__ W,
                      unsigned short* __restrict__ WeT) {
    int gid = blockIdx.x * blockDim.x + threadIdx.x;   // 0..524287
    int h = gid >> 10;
    int e = gid & (EDIM - 1);
    WeT[(size_t)h * EDIM + e] = f2bf(W[(size_t)(HDIM + e) * HDIM + h]);
}

// ---------------------------------------------------------------------------
// logits[b,s] = sum_h tanh( hid_proj[b,h] + (enc[b,s,:] . We[:,h]) ) * v[h]
// Workgroup: one batch b, 64 consecutive s rows. A tile staged in LDS as bf16
// in a K-block-swizzled layout so each lane's A fragment is 32 contiguous
// bytes (2x ds_load_b128). GEMM via v_wmma_f32_16x16x32_bf16, 4-way N-tile
// register blocking, epilogue fused.
__global__ void k_logits(const float* __restrict__ enc,
                         const unsigned short* __restrict__ WeT,
                         const float* __restrict__ hidp,
                         const float* __restrict__ v,
                         float* __restrict__ logits) {
    extern __shared__ unsigned short lds16[];   // MTILE x EDIM bf16 (128 KB)
    const int b   = blockIdx.y;
    const int s0  = blockIdx.x * MTILE;
    const int tid = threadIdx.x;

    // Stage + convert: 64 x 1024 fp32 -> bf16 LDS.
    // Swizzle: within each 32-wide K chunk, swap 8-element blocks 1<->2 so a
    // lane-half's 16 fragment elements are contiguous:
    //   positions [0..15]  = K {0..7, 16..23}   (half 0)
    //   positions [16..31] = K {8..15, 24..31}  (half 1)
    const size_t encBase = ((size_t)b * SEQ + s0) * EDIM;
    for (int idx = tid * 4; idx < MTILE * EDIM; idx += LG_BLOCK * 4) {
        int row = idx >> 10;
        int col = idx & (EDIM - 1);
        float4 f = *(const float4*)(enc + encBase + ((size_t)row << 10) + col);
        union { unsigned short s[4]; uint2 u; } pk;
        pk.s[0] = f2bf(f.x); pk.s[1] = f2bf(f.y);
        pk.s[2] = f2bf(f.z); pk.s[3] = f2bf(f.w);
        int blk  = (col >> 3) & 3;
        int nblk = ((blk & 1) << 1) | (blk >> 1);      // 0,1,2,3 -> 0,2,1,3
        int dst  = (row << 10) + (col & ~31) + (nblk << 3) + (col & 7);
        *(uint2*)(lds16 + dst) = pk.u;
    }
    __syncthreads();

    const int wave  = tid >> 5;
    const int lane  = tid & 31;
    const int half  = lane >> 4;      // K sub-range / C row offset
    const int n     = lane & 15;      // A row within tile == B/C column
    const int mrow0 = wave * 16;      // this wave's 16-row block

    float lg[8];
#pragma unroll
    for (int r = 0; r < 8; ++r) lg[r] = 0.f;

    // Per-lane base pointers (fragment-contiguous thanks to the swizzle)
    const unsigned short* ap = lds16 + (size_t)(mrow0 + n) * EDIM + 16 * half;

    for (int ng = 0; ng < HDIM / (16 * NBLK); ++ng) {     // 8 groups of 4 n-tiles
        v8f acc[NBLK];
#pragma unroll
        for (int t = 0; t < NBLK; ++t) acc[t] = (v8f){};

        const unsigned short* bp =
            WeT + (size_t)(ng * (16 * NBLK) + n) * EDIM + 8 * half;

#pragma unroll 2
        for (int kb = 0; kb < EDIM; kb += 32) {
            union { v16bf v; uint4 q[2]; } af;
            const uint4* aq = (const uint4*)(ap + kb);
            af.q[0] = aq[0];
            af.q[1] = aq[1];

#pragma unroll
            for (int t = 0; t < NBLK; ++t) {
                union { v16bf v; uint4 q[2]; } bfm;
                const uint4* bq = (const uint4*)(bp + (size_t)t * 16 * EDIM + kb);
                bfm.q[0] = bq[0];
                bfm.q[1] = bq[2];
                acc[t] = __builtin_amdgcn_wmma_f32_16x16x32_bf16(
                    false, af.v, false, bfm.v, (short)0, acc[t], false, false);
            }
        }

        // Fused epilogue: energy = tanh(acc + hid_proj), partial logit += e * v
#pragma unroll
        for (int t = 0; t < NBLK; ++t) {
            int hcol = ng * (16 * NBLK) + t * 16 + n;
            float hp = hidp[b * HDIM + hcol];
            float vv = v[hcol];
#pragma unroll
            for (int r = 0; r < 8; ++r) {
                float e = tanhf(acc[t][r] + hp);   // row m=r+8*half, col hcol
                lg[r] += e * vv;
            }
        }
    }

    // Sum over the 16 lanes of each half (columns) -> per-row logits
#pragma unroll
    for (int r = 0; r < 8; ++r) {
        lg[r] += __shfl_xor(lg[r], 1, 32);
        lg[r] += __shfl_xor(lg[r], 2, 32);
        lg[r] += __shfl_xor(lg[r], 4, 32);
        lg[r] += __shfl_xor(lg[r], 8, 32);
    }
    if (n == 0) {
#pragma unroll
        for (int r = 0; r < 8; ++r) {
            int m = r + 8 * half;
            logits[(size_t)b * SEQ + s0 + mrow0 + m] = lg[r];
        }
    }
}

// ---------------------------------------------------------------------------
// In-place softmax over S for each batch (one block per batch).
__global__ void k_softmax(float* __restrict__ logits) {
    const int b = blockIdx.x;
    float* lp = logits + (size_t)b * SEQ;
    const int tid = threadIdx.x;   // 256

    __shared__ float sred[8];

    float mx = -3.402823466e38f;
    for (int i = tid; i < SEQ; i += 256) mx = fmaxf(mx, lp[i]);
#pragma unroll
    for (int o = 16; o >= 1; o >>= 1) mx = fmaxf(mx, __shfl_xor(mx, o, 32));
    if ((tid & 31) == 0) sred[tid >> 5] = mx;
    __syncthreads();
    float gmax = sred[0];
#pragma unroll
    for (int i = 1; i < 8; ++i) gmax = fmaxf(gmax, sred[i]);
    __syncthreads();

    float sum = 0.f;
    for (int i = tid; i < SEQ; i += 256) sum += expf(lp[i] - gmax);
#pragma unroll
    for (int o = 16; o >= 1; o >>= 1) sum += __shfl_xor(sum, o, 32);
    if ((tid & 31) == 0) sred[tid >> 5] = sum;
    __syncthreads();
    float gsum = 0.f;
#pragma unroll
    for (int i = 0; i < 8; ++i) gsum += sred[i];

    float inv = 1.0f / gsum;
    for (int i = tid; i < SEQ; i += 256) lp[i] = expf(lp[i] - gmax) * inv;
}

// ---------------------------------------------------------------------------
// part[b,c,e] = sum over 128 s rows of attn[b,s] * enc[b,s,e]
__global__ void k_ctx_partial(const float* __restrict__ enc,
                              const float* __restrict__ attn,
                              float* __restrict__ part) {
    const int b   = blockIdx.y;    // 0..15
    const int c   = blockIdx.x;    // 0..31
    const int col = threadIdx.x * 4;
    float4 acc = make_float4(0.f, 0.f, 0.f, 0.f);
    const int s0 = c * 128;
    for (int s = s0; s < s0 + 128; ++s) {
        float w = attn[(size_t)b * SEQ + s];
        const float* rp = enc + (((size_t)b * SEQ + s) << 10);
        __builtin_prefetch(rp + EDIM + col, 0, 1);   // -> global_prefetch_b8
        float4 ev = *(const float4*)(rp + col);
        acc.x += w * ev.x; acc.y += w * ev.y;
        acc.z += w * ev.z; acc.w += w * ev.w;
    }
    *(float4*)(part + (((size_t)b * 32 + c) << 10) + col) = acc;
}

// context[b,e] = sum_c part[b,c,e]   (fixed-order -> deterministic)
__global__ void k_ctx_reduce(const float* __restrict__ part,
                             float* __restrict__ out) {
    int gid = blockIdx.x * blockDim.x + threadIdx.x;  // 0..16383
    int b = gid >> 10;
    int e = gid & (EDIM - 1);
    float s = 0.f;
#pragma unroll 4
    for (int c = 0; c < 32; ++c) s += part[(((size_t)b * 32 + c) << 10) + e];
    out[gid] = s;
}

// ---------------------------------------------------------------------------
extern "C" void kernel_launch(void* const* d_in, const int* in_sizes, int n_in,
                              void* d_out, int out_size, void* d_ws, size_t ws_size,
                              hipStream_t stream) {
    (void)in_sizes; (void)n_in; (void)out_size; (void)ws_size;
    const float* hidden = (const float*)d_in[0];   // (16, 512)
    const float* enc    = (const float*)d_in[1];   // (16, 4096, 1024)
    const float* W      = (const float*)d_in[2];   // (1536, 512)
    const float* bias   = (const float*)d_in[3];   // (512,)
    const float* v      = (const float*)d_in[4];   // (512, 1)
    float* out = (float*)d_out;                    // (16, 1, 1024)

    char* ws = (char*)d_ws;
    float*          hidp   = (float*)ws;                                   // 32 KB
    unsigned short* WeT    = (unsigned short*)(ws + (32u << 10));          // 1 MB
    float*          logits = (float*)(ws + (32u << 10) + (1u << 20));      // 256 KB
    float*          part   = (float*)(ws + (32u << 10) + (1u << 20) + (256u << 10)); // 2 MB

    k_hidproj<<<dim3((BATCH * HDIM) / 256), 256, 0, stream>>>(hidden, W, bias, hidp);
    k_wet<<<dim3((HDIM * EDIM) / 256), 256, 0, stream>>>(W, WeT);
    k_logits<<<dim3(SEQ / MTILE, BATCH), LG_BLOCK,
               MTILE * EDIM * sizeof(unsigned short), stream>>>(enc, WeT, hidp, v, logits);
    k_softmax<<<dim3(BATCH), 256, 0, stream>>>(logits);
    k_ctx_partial<<<dim3(32, BATCH), 256, 0, stream>>>(enc, logits, part);
    k_ctx_reduce<<<dim3((BATCH * EDIM) / 256), 256, 0, stream>>>(part, out);
}
